// LPKTNet_82987358093629
// MI455X (gfx1250) — matrix operand order repaired and checked by reading, not compiled, not run on hardware
//
#include <hip/hip_runtime.h>
#include <hip/hip_bf16.h>

#define BSZ      64
#define SEQ      50
#define NSKILL   512
#define DK       128
#define DA       50
#define DE       128
#define IT_NUM   1440
#define HSTRIDE  132   // padded h row stride (words); 132 mod 64 = 4 -> conflict-free-ish

typedef __attribute__((ext_vector_type(16))) __bf16        v16bf;
typedef __attribute__((ext_vector_type(8)))  float         v8f;
typedef __attribute__((ext_vector_type(8)))  unsigned int  v8u;
typedef __attribute__((ext_vector_type(4)))  unsigned int  v4u;

union ABfrag  { v8u u; v16bf b; };
union Bfrag2  { v4u u4[2]; v16bf b; };

// pack two floats into a dword of two bf16 (v_cvt_pk_bf16_f32)
__device__ __forceinline__ unsigned int pack_bf16(float lo, float hi) {
    union { __bf16 b[2]; unsigned int u; } p;
    p.b[0] = (__bf16)lo;
    p.b[1] = (__bf16)hi;
    return p.u;
}

__device__ __forceinline__ float sigmoidf(float x) {
    return 1.0f / (1.0f + __expf(-x));
}

// ---------------------------------------------------------------------------
// Kernel 1: learning[b,s,:] = concat(ex, at, ans) @ W_le.T + b_le
// ---------------------------------------------------------------------------
__global__ __launch_bounds__(128)
void lpkt_learning_kernel(const float* __restrict__ E_emb,
                          const float* __restrict__ AT_emb,
                          const float* __restrict__ W_le,
                          const float* __restrict__ b_le,
                          const float* __restrict__ answer_value,
                          const int*   __restrict__ exercise_id,
                          const int*   __restrict__ ans_time,
                          float*       __restrict__ learning_ws) {
    const int bs = blockIdx.x;          // 0 .. B*S-1
    const int o  = threadIdx.x;         // 0 .. 127
    const int e  = exercise_id[bs];
    const int a  = ans_time[bs];
    const float av = answer_value[bs];

    const float* ex = E_emb  + (size_t)e * DE;
    const float* at = AT_emb + (size_t)a * DK;
    const float* w  = W_le   + (size_t)o * (DE + DK + DA);

    float s = b_le[o];
    #pragma unroll 4
    for (int i = 0; i < DE; ++i) s += ex[i] * w[i];
    #pragma unroll 4
    for (int i = 0; i < DK; ++i) s += at[i] * w[DE + i];
    float wsum = 0.0f;
    #pragma unroll 5
    for (int i = 0; i < DA; ++i) wsum += w[DE + DK + i];
    s += av * wsum;

    learning_ws[(size_t)bs * DK + o] = s;
}

// ---------------------------------------------------------------------------
// Kernel 2: full recurrence, one workgroup per batch element.
// h[512,128] f32 lives in LDS (row stride 132 for bank spread) for all 49
// steps; per-step GEMM h @ W_f1^T via v_wmma_f32_16x16x32_bf16.
// ---------------------------------------------------------------------------
__global__ __launch_bounds__(256)
void lpkt_main_kernel(const float* __restrict__ q_matrix,
                      const float* __restrict__ h0,
                      const float* __restrict__ E_emb,
                      const float* __restrict__ IT_emb,
                      const float* __restrict__ W_l, const float* __restrict__ b_l,
                      const float* __restrict__ W_g, const float* __restrict__ b_g,
                      const float* __restrict__ W_f, const float* __restrict__ b_f,
                      const float* __restrict__ W_p, const float* __restrict__ b_p,
                      const int*   __restrict__ exercise_id,
                      const int*   __restrict__ interval_time,
                      const float* __restrict__ learning_ws,
                      float*       __restrict__ pred) {
    extern __shared__ __align__(32) char smem_raw[];
    float*        h     = (float*)smem_raw;                     // 512*132 f32 (264 KB)
    unsigned int* wf1   = (unsigned int*)(smem_raw + 270336);   // 8192 dwords (32 KB)
    float*        g     = (float*)(smem_raw + 303104);          // gate_in [512]
    float*        qe    = g   + 512;                            // q_e    [512]
    float*        qn    = qe  + 512;                            // q_next [512]
    float*        LG    = qn  + 512;                            // [128]
    float*        biasf = LG  + 128;                            // [128]
    float*        zl    = biasf + 128;                          // [128]
    float*        zg    = zl  + 128;                            // [128]
    float*        htl   = zg  + 128;                            // h_tilde [128]
    float*        pc1   = htl + 128;                            // pred part [128]
    float*        red   = pc1 + 128;                            // [8]

    const int b    = blockIdx.x;
    const int tid  = threadIdx.x;
    const int lane = tid & 31;
    const int wave = tid >> 5;

    // ---- init: h <- h0 (padded rows) ----
    for (int i = tid; i < NSKILL * DK; i += 256)
        h[(i >> 7) * HSTRIDE + (i & 127)] = h0[i];

    // ---- build bf16 B-fragments of W_f[:, 0:128]^T ----
    // layout: [kt(4)][nt(8)][chunk(2)][lane(32)][dword(4)]  (chunked for 2-way-max banks)
// per 16x16x32 tile: lanes 0-15 hold K = kt*32 + {0..7,16..23} (dword v -> 2 consecutive K),
//                    lanes 16-31 hold K = kt*32 + {8..15,24..31}; N = nt*16 + lane%16.
    for (int idx = tid; idx < 4 * 8 * 2 * 32 * 4; idx += 256) {
        const int vin = idx & 3;
        const int l   = (idx >> 2) & 31;
        const int ch  = (idx >> 7) & 1;
        const int nt  = (idx >> 8) & 7;
        const int kt  = idx >> 11;
        const int v   = ch * 4 + vin;
        const int n   = nt * 16 + (l & 15);
        const int kb  = kt * 32 + ((l >= 16) ? 8 : 0) + ch * 16 + 2 * vin;
        wf1[idx] = pack_bf16(W_f[(size_t)n * 384 + kb], W_f[(size_t)n * 384 + kb + 1]);
    }

    // ---- q_e[:,0], learning_pre = 0, h_tilde0 = q_e0 . h0 ----
    {
        const int e0 = exercise_id[b * SEQ];
        for (int i = tid; i < NSKILL; i += 256) qe[i] = q_matrix[(size_t)e0 * NSKILL + i];
        if (tid < DK) g[tid] = 0.0f;                    // learning_pre
        __syncthreads();
        if (tid < DK) {
            float s = 0.0f;
            for (int n = 0; n < NSKILL; ++n) s += qe[n] * h[n * HSTRIDE + tid];
            g[384 + tid] = s;                           // h_tilde_pre
        }
        if (tid == 0) pred[b * SEQ] = 0.0f;
    }
    __syncthreads();

    for (int t = 0; t < SEQ - 1; ++t) {
        const int e_t = exercise_id[b * SEQ + t];
        const int e_n = exercise_id[b * SEQ + t + 1];
        int itv = interval_time[b * SEQ + t] / 60;
        itv = (itv < 0) ? 0 : ((itv > IT_NUM - 1) ? IT_NUM - 1 : itv);

        // ---- gather: it_t, l_t, q_e, q_next; zero h_tilde accumulator ----
        if (tid < DK) {
            g[128 + tid] = IT_emb[(size_t)itv * DK + tid];                 // it_t
            g[256 + tid] = learning_ws[((size_t)b * SEQ + t) * DK + tid];  // l_t
            htl[tid] = 0.0f;
        }
        for (int i = tid; i < NSKILL; i += 256) {
            qe[i] = q_matrix[(size_t)e_t * NSKILL + i];
            qn[i] = q_matrix[(size_t)e_n * NSKILL + i];
        }
        __syncthreads();

        // ---- gate matvecs: zl = gate_in @ W_l^T, zg = gate_in @ W_g^T ----
        {
            const int o = tid & 127;
            const float*  W  = (tid < 128) ? W_l : W_g;
            const float*  bb = (tid < 128) ? b_l : b_g;
            const float4* Wr = (const float4*)(W + (size_t)o * 512);
            const float4* gr = (const float4*)g;
            float s = bb[o];
            #pragma unroll 8
            for (int i = 0; i < 128; ++i) {
                const float4 w = Wr[i];
                const float4 x = gr[i];
                s += w.x * x.x + w.y * x.y + w.z * x.z + w.w * x.w;
            }
            if (tid < 128) zl[o] = s; else zg[o] = s;
        }
        __syncthreads();

        // ---- LG (threads 0-127) ; pc1 = b_p + e_next @ W_p[:, :128] (128-255) ----
        if (tid < 128) {
            const float lgv = tanhf(zl[tid]);
            const float gl  = sigmoidf(zg[tid]);
            LG[tid] = gl * (lgv + 1.0f) * 0.5f;
        } else {
            const int o = tid - 128;
            const float* er = E_emb + (size_t)e_n * DE;
            float s = b_p[o];
            #pragma unroll 8
            for (int i = 0; i < DK; ++i) s += er[i] * W_p[(size_t)o * 256 + i];
            pc1[o] = s;
        }
        __syncthreads();

        // ---- bias_f = b_f + LG @ W_f2^T + it @ W_f3^T ----
        if (tid < 128) {
            const int o = tid;
            float s = b_f[o];
            const float* wr = W_f + (size_t)o * 384;
            #pragma unroll 8
            for (int i = 0; i < DK; ++i)
                s += LG[i] * wr[128 + i] + g[128 + i] * wr[256 + i];
            biasf[o] = s;
        }
        __syncthreads();

        // ---- prefetch next step's gather rows into cache (global_prefetch_b8)
        if (t + 2 < SEQ && tid == 0) {
            const int e1 = exercise_id[b * SEQ + t + 1];
            const int e2 = exercise_id[b * SEQ + t + 2];
            int itv1 = interval_time[b * SEQ + t + 1] / 60;
            itv1 = (itv1 < 0) ? 0 : ((itv1 > IT_NUM - 1) ? IT_NUM - 1 : itv1);
            __builtin_prefetch(q_matrix + (size_t)e1 * NSKILL, 0, 2);
            __builtin_prefetch(q_matrix + (size_t)e2 * NSKILL, 0, 2);
            __builtin_prefetch(E_emb + (size_t)e2 * DE, 0, 2);
            __builtin_prefetch(IT_emb + (size_t)itv1 * DK, 0, 2);
            __builtin_prefetch(learning_ws + ((size_t)b * SEQ + t + 1) * DK, 0, 2);
        }

        // ---- main GEMM (WMMA bf16) + fused state update + h_tilde reduce ----
        {
            float htp[8];
            #pragma unroll
            for (int q = 0; q < 8; ++q) htp[q] = 0.0f;

            const int koffBase = (lane >= 16) ? 8 : 0;

            #pragma unroll 1
            for (int mi = 0; mi < 4; ++mi) {
                const int mt = wave * 4 + mi;
                const int m  = mt * 16 + (lane & 15);
                const float* hrow = h + (size_t)m * HSTRIDE;

                // A fragments for all 4 k-slices (f32 -> bf16 native converts)
                ABfrag afr[4];
                #pragma unroll
                for (int kt = 0; kt < 4; ++kt) {
                    const int k0 = kt * 32 + koffBase;
                    const float4 p0 = *(const float4*)(hrow + k0);
                    const float4 p1 = *(const float4*)(hrow + k0 + 4);
                    const float4 p2 = *(const float4*)(hrow + k0 + 16);
                    const float4 p3 = *(const float4*)(hrow + k0 + 20);
                    v16bf a;
                    a[0]  = (__bf16)p0.x; a[1]  = (__bf16)p0.y;
                    a[2]  = (__bf16)p0.z; a[3]  = (__bf16)p0.w;
                    a[4]  = (__bf16)p1.x; a[5]  = (__bf16)p1.y;
                    a[6]  = (__bf16)p1.z; a[7]  = (__bf16)p1.w;
                    a[8]  = (__bf16)p2.x; a[9]  = (__bf16)p2.y;
                    a[10] = (__bf16)p2.z; a[11] = (__bf16)p2.w;
                    a[12] = (__bf16)p3.x; a[13] = (__bf16)p3.y;
                    a[14] = (__bf16)p3.z; a[15] = (__bf16)p3.w;
                    afr[kt].b = a;
                }

                v8f acc[8];
                #pragma unroll
                for (int nt = 0; nt < 8; ++nt)
                    acc[nt] = (v8f){0.f,0.f,0.f,0.f,0.f,0.f,0.f,0.f};

                #pragma unroll
                for (int kt = 0; kt < 4; ++kt) {
                    #pragma unroll
                    for (int nt = 0; nt < 8; ++nt) {
                        Bfrag2 bf;
                        const unsigned int* base =
                            wf1 + (((kt * 8 + nt) * 2) << 7) + (lane << 2);
                        bf.u4[0] = *(const v4u*)(base);
                        bf.u4[1] = *(const v4u*)(base + 128);
                        acc[nt] = __builtin_amdgcn_wmma_f32_16x16x32_bf16(
                            false, afr[kt].b, false, bf.b, (short)0, acc[nt], false, false);
                    }
                }

                // fused update of this wave's 16 rows:
                // D layout: VGPR r -> M = mt*16 + r (+8 for lanes>=16), N = nt*16 + lane%16
                const int rowbase = mt * 16 + ((lane >= 16) ? 8 : 0);
                const int nc      = lane & 15;
                #pragma unroll
                for (int nt = 0; nt < 8; ++nt) {
                    const int   N   = nt * 16 + nc;
                    const float bfv = biasf[N];
                    const float lgN = LG[N];
                    float part = 0.0f;
                    #pragma unroll
                    for (int r = 0; r < 8; ++r) {
                        const int   M  = rowbase + r;
                        const float z  = acc[nt][r] + bfv;
                        const float gf = sigmoidf(z);
                        const float hv = h[(size_t)M * HSTRIDE + N];
                        const float hn = qe[M] * lgN + gf * hv;
                        h[(size_t)M * HSTRIDE + N] = hn;
                        part += qn[M] * hn;
                    }
                    htp[nt] += part;
                }
            }

            // h_tilde partial sums -> LDS (ds_add_f32)
            #pragma unroll
            for (int nt = 0; nt < 8; ++nt)
                atomicAdd(&htl[nt * 16 + (lane & 15)], htp[nt]);
        }
        __syncthreads();

        // ---- prediction: y = mean(sigmoid(pc1 + h_tilde @ W_p[:,128:256]^T)) ----
        if (tid < 128) {
            const int o = tid;
            float s = pc1[o];
            const float* wr = W_p + (size_t)o * 256 + 128;
            #pragma unroll 8
            for (int i = 0; i < DK; ++i) s += htl[i] * wr[i];
            float y = sigmoidf(s);
            #pragma unroll
            for (int off = 16; off > 0; off >>= 1) y += __shfl_down(y, off, 32);
            if (lane == 0) red[wave] = y;
        }
        __syncthreads();
        if (tid == 0)
            pred[b * SEQ + t + 1] = (red[0] + red[1] + red[2] + red[3]) * (1.0f / 128.0f);

        // rotate carried state: learning_pre <- l_t ; h_tilde_pre <- h_tilde
        if (tid < DK) {
            g[tid]       = g[256 + tid];
            g[384 + tid] = htl[tid];
        }
        __syncthreads();
    }
}

extern "C" void kernel_launch(void* const* d_in, const int* in_sizes, int n_in,
                              void* d_out, int out_size, void* d_ws, size_t ws_size,
                              hipStream_t stream) {
    const float* answer_value  = (const float*)d_in[0];
    const float* q_matrix      = (const float*)d_in[1];
    const float* h0            = (const float*)d_in[2];
    const float* E_emb         = (const float*)d_in[3];
    const float* AT_emb        = (const float*)d_in[4];
    const float* IT_emb        = (const float*)d_in[5];
    const float* W_le          = (const float*)d_in[6];
    const float* b_le          = (const float*)d_in[7];
    const float* W_l           = (const float*)d_in[8];
    const float* b_l           = (const float*)d_in[9];
    const float* W_g           = (const float*)d_in[10];
    const float* b_g           = (const float*)d_in[11];
    const float* W_f           = (const float*)d_in[12];
    const float* b_f           = (const float*)d_in[13];
    const float* W_p           = (const float*)d_in[14];
    const float* b_p           = (const float*)d_in[15];
    const int*   exercise_id   = (const int*)d_in[16];
    // d_in[17] skill_id, d_in[18] stu_id are unused by the reference
    const int*   ans_time      = (const int*)d_in[19];
    const int*   interval_time = (const int*)d_in[20];

    float* pred        = (float*)d_out;
    float* learning_ws = (float*)d_ws;   // B*S*DK floats = 1.6 MB

    lpkt_learning_kernel<<<BSZ * SEQ, 128, 0, stream>>>(
        E_emb, AT_emb, W_le, b_le, answer_value, exercise_id, ans_time, learning_ws);

    // LDS: h 264KB (padded) + wf1 32KB + gate/aux ~9.3KB  (fits 320KB WGP LDS)
    const size_t smem = 512 * HSTRIDE * 4 + 32768
                      + (512 + 512 + 512 + 128 * 6 + 8) * sizeof(float);
    (void)hipFuncSetAttribute((const void*)lpkt_main_kernel,
                              hipFuncAttributeMaxDynamicSharedMemorySize, (int)smem);
    lpkt_main_kernel<<<BSZ, 256, smem, stream>>>(
        q_matrix, h0, E_emb, IT_emb, W_l, b_l, W_g, b_g, W_f, b_f, W_p, b_p,
        exercise_id, interval_time, learning_ws, pred);
}
